// PHI_37933151158977
// MI455X (gfx1250) — compile-verified, matrix operand order checked by reading, not run on hardware
//
#include <hip/hip_runtime.h>
#include <hip/hip_bf16.h>
#include <stdint.h>

typedef __attribute__((ext_vector_type(16))) __bf16 v16bf;
typedef __attribute__((ext_vector_type(8)))  float  v8f;
typedef __attribute__((ext_vector_type(4)))  unsigned int u32x4;
typedef __attribute__((ext_vector_type(8)))  int i32x8;
typedef __attribute__((ext_vector_type(4)))  int i32x4;

#define LAYERS 2
#define NH     24
#define NKV    8
#define GQA    3
#define HD     128
#define ROTD   96
#define RHALF  48
#define HIDN   3072
#define FFN    8192
#define VOCAB  32000
#define SEQ    1024
#define QPN    (NH*HD)        // 3072
#define KPN    (NKV*HD)       // 1024
#define QKVN   (QPN + 2*KPN)  // 5120
#define EPSV   1e-6f
#define QKSCALE 0.2973017787506803f   // 128^(-0.25)

#define BM 128
#define BN 128
#define LDSTR 40   // padded LDS row stride (elements): 32 data + 8 pad -> bank-conflict free

#define HAVE_TDM __has_builtin(__builtin_amdgcn_tensor_load_to_lds)

__device__ __forceinline__ unsigned short f2bf(float f) {
    union { float f; unsigned int u; } v; v.f = f;
    unsigned int u = v.u;
    unsigned int r = u + 0x7FFFu + ((u >> 16) & 1u);
    return (unsigned short)(r >> 16);
}
__device__ __forceinline__ float bf2f(unsigned short s) {
    union { float f; unsigned int u; } v; v.u = ((unsigned int)s) << 16;
    return v.f;
}
// 32-byte fragment load as two 16B chunks (only 16B alignment required)
__device__ __forceinline__ v16bf ldsFrag(const unsigned short* p) {
    union { v16bf v; uint4 q[2]; } u;
    u.q[0] = *(const uint4*)p;
    u.q[1] = *(const uint4*)(p + 8);
    return u.v;
}

// ---------------------------------------------------------------- embedding
__global__ __launch_bounds__(256) void embed_kernel(
    const int* __restrict__ ids, const unsigned char* __restrict__ eq,
    const float* __restrict__ esc, const float* __restrict__ ez,
    float* __restrict__ h)
{
    int s = blockIdx.x;
    int id = ids[s];
    float sc = esc[id], z = ez[id];
    const unsigned char* row = eq + (size_t)id * HIDN;
    for (int c = threadIdx.x; c < HIDN; c += 256)
        h[(size_t)s * HIDN + c] = (float)row[c] * sc + z;
}

// ---------------------------------------------------------------- RMSNorm -> bf16
__global__ __launch_bounds__(256) void rms_kernel(
    const float* __restrict__ x, const float* __restrict__ w,
    unsigned short* __restrict__ out)
{
    __shared__ float red[8];
    int row = blockIdx.x;
    const float* xr = x + (size_t)row * HIDN;
    float ss = 0.f;
    for (int c = threadIdx.x; c < HIDN; c += 256) { float t = xr[c]; ss += t * t; }
    #pragma unroll
    for (int d = 1; d < 32; d <<= 1) ss += __shfl_xor(ss, d, 32);
    if ((threadIdx.x & 31) == 0) red[threadIdx.x >> 5] = ss;
    __syncthreads();
    if (threadIdx.x < 8) {
        float v = red[threadIdx.x];
        #pragma unroll
        for (int d = 1; d < 8; d <<= 1) v += __shfl_xor(v, d, 32);
        if (threadIdx.x == 0) red[0] = v;
    }
    __syncthreads();
    float rinv = rsqrtf(red[0] / (float)HIDN + EPSV);
    for (int c = threadIdx.x; c < HIDN; c += 256)
        out[(size_t)row * HIDN + c] = f2bf(w[c] * xr[c] * rinv);
}

// ---------------------------------------------------------------- WMMA GEMM
// C[M,N] = (RES ? RES : 0) + A_bf16[M,K] * W_f32[N,K]^T
// 256 threads (8 wave32). Tile 128x128, K-step 32, double-buffered LDS.
// A tile moved by the Tensor Data Mover (TDM) with LDS padding; W tile
// converted fp32->bf16 on VALU (co-executes with WMMA on the matrix pipes).
// Grid: x = M-tile (fast) so co-resident blocks share the W stream in L2.
__global__ __launch_bounds__(256) void gemm_bf16_kernel(
    const unsigned short* __restrict__ A,
    const float* __restrict__ W,
    const float* __restrict__ RES,
    float* __restrict__ C,
    int M, int N, int K)
{
    __shared__ unsigned short lA[2][BM * LDSTR];
    __shared__ unsigned short lB[2][BN * LDSTR];
    const int tid  = threadIdx.x;
    const int lane = tid & 31;
    const int wave = tid >> 5;
    const int wm = (wave & 3) * 32;     // 4 M-groups of 32
    const int wn = (wave >> 2) * 64;    // 2 N-groups of 64
    const int bM = blockIdx.x * BM;
    const int bN = blockIdx.y * BN;
    const int laneHalf = lane >> 4;
    const int laneMod  = lane & 15;
    const int wrow = tid >> 1, wseg = tid & 1;

    v8f acc[2][4] = {};

    auto issueA = [&](int k0, int bufIdx) {
#if HAVE_TDM
        if (wave == 0) {
            unsigned long long ga = (unsigned long long)(A + (size_t)bM * K + k0);
            unsigned lds = (unsigned)(unsigned long long)&lA[bufIdx][0];
            u32x4 g0;
            g0[0] = 1u;                                          // count=1 (valid D#)
            g0[1] = lds;                                         // lds_addr
            g0[2] = (unsigned)(ga & 0xFFFFFFFFu);                // global_addr[31:0]
            g0[3] = (unsigned)((ga >> 32) & 0x1FFFFFFu) | (2u << 30); // addr[56:32] | type=2
            i32x8 g1;
            // data_size=2B(code1), pad_enable, pad_interval=16dw(code3), pad_amount=4dw(code3)
            g1[0] = (int)((1u << 16) | (1u << 20) | (3u << 22) | (3u << 25));
            g1[1] = (int)(((unsigned)K & 0xFFFFu) << 16);        // tensor_dim0[15:0]
            g1[2] = (int)((((unsigned)K >> 16) & 0xFFFFu) |
                          (((unsigned)M & 0xFFFFu) << 16));      // dim0[31:16] | dim1[15:0]
            g1[3] = (int)((((unsigned)M >> 16) & 0xFFFFu) |
                          (32u << 16));                          // dim1[31:16] | tile_dim0=32
            g1[4] = (int)BM;                                     // tile_dim1=128, tile_dim2=0
            g1[5] = (int)K;                                      // dim0_stride[31:0]
            g1[6] = 0; g1[7] = 0;
            i32x4 z4 = {0, 0, 0, 0};
#if __clang_major__ >= 23
            i32x8 z8 = {0, 0, 0, 0, 0, 0, 0, 0};
            __builtin_amdgcn_tensor_load_to_lds(g0, g1, z4, z4, z8, 0);
#else
            __builtin_amdgcn_tensor_load_to_lds(g0, g1, z4, z4, 0);
#endif
        }
#else
        int row = tid >> 1, seg = tid & 1;
        const unsigned short* src = A + (size_t)(bM + row) * K + k0 + seg * 16;
        uint4 a0 = *(const uint4*)src;
        uint4 a1 = *(const uint4*)(src + 8);
        *(uint4*)&lA[bufIdx][row * LDSTR + seg * 16] = a0;
        *(uint4*)&lA[bufIdx][row * LDSTR + seg * 16 + 8] = a1;
#endif
    };
    auto waitA = [&]() {
#if HAVE_TDM
        if (wave == 0) __builtin_amdgcn_s_wait_tensorcnt(0);
#endif
    };
    auto loadWregs = [&](int k0, float4* r) {
        const float* src = W + (size_t)(bN + wrow) * K + k0 + wseg * 16;
        r[0] = *(const float4*)(src);
        r[1] = *(const float4*)(src + 4);
        r[2] = *(const float4*)(src + 8);
        r[3] = *(const float4*)(src + 12);
    };
    auto storeW = [&](int bufIdx, const float4* r) {
        unsigned short* dst = &lB[bufIdx][wrow * LDSTR + wseg * 16];
        #pragma unroll
        for (int j = 0; j < 4; ++j) {
            dst[4 * j + 0] = f2bf(r[j].x); dst[4 * j + 1] = f2bf(r[j].y);
            dst[4 * j + 2] = f2bf(r[j].z); dst[4 * j + 3] = f2bf(r[j].w);
        }
    };

    float4 wr[4];
    issueA(0, 0);
    loadWregs(0, wr);
    waitA();
    storeW(0, wr);
    __syncthreads();

    int buf = 0;
    for (int k0 = 0; k0 < K; k0 += 32) {
        const int nxt = k0 + 32;
        const bool more = nxt < K;
        if (more) { issueA(nxt, buf ^ 1); loadWregs(nxt, wr); }

        v16bf af[2], bfr[4];
        #pragma unroll
        for (int mi = 0; mi < 2; ++mi)
            af[mi] = ldsFrag(&lA[buf][(wm + mi * 16 + laneMod) * LDSTR + laneHalf * 16]);
        #pragma unroll
        for (int ni = 0; ni < 4; ++ni)
            bfr[ni] = ldsFrag(&lB[buf][(wn + ni * 16 + laneMod) * LDSTR + laneHalf * 16]);
        #pragma unroll
        for (int mi = 0; mi < 2; ++mi)
            #pragma unroll
            for (int ni = 0; ni < 4; ++ni)
                acc[mi][ni] = __builtin_amdgcn_wmma_f32_16x16x32_bf16(
                    false, af[mi], false, bfr[ni], (short)0, acc[mi][ni], false, false);

        if (more) {
            waitA();
            storeW(buf ^ 1, wr);
        }
        __syncthreads();
        buf ^= 1;
    }

    #pragma unroll
    for (int mi = 0; mi < 2; ++mi)
        #pragma unroll
        for (int ni = 0; ni < 4; ++ni)
            #pragma unroll
            for (int r = 0; r < 8; ++r) {
                int m  = r + 8 * laneHalf;
                int gr = bM + wm + mi * 16 + m;
                int gc = bN + wn + ni * 16 + laneMod;
                size_t idx = (size_t)gr * N + gc;
                float v = acc[mi][ni][r];
                if (RES) v += RES[idx];
                C[idx] = v;
            }
}

// ---------------------------------------------------------------- RoPE + KV write
__global__ __launch_bounds__(256) void rope_kv_kernel(
    const float* __restrict__ qkv,
    const float* __restrict__ cosE, const float* __restrict__ sinE,
    unsigned short* __restrict__ qb,   // [NH,SEQ,HD] bf16
    unsigned short* __restrict__ kbuf, // [NKV,SEQ,HD] bf16
    unsigned short* __restrict__ vT,   // [NKV,HD,SEQ] bf16
    float* __restrict__ keys_out,      // [NKV,HD,SEQ] fp32
    float* __restrict__ vals_out)      // [NKV,SEQ,HD] fp32
{
    int s = blockIdx.x;
    const float* row = qkv + (size_t)s * QKVN;
    for (int t = threadIdx.x; t < QKVN; t += 256) {
        if (t < QPN) {
            int hh = t / HD, d = t % HD;
            float val = row[t] * QKSCALE;
            if (d < ROTD) {
                float c = cosE[s * ROTD + d], sn = sinE[s * ROTD + d];
                float partner = (d < RHALF) ? -row[hh * HD + d + RHALF]
                                            :  row[hh * HD + d - RHALF];
                val = val * c + partner * QKSCALE * sn;
            }
            qb[(size_t)hh * SEQ * HD + (size_t)s * HD + d] = f2bf(val);
        } else if (t < QPN + KPN) {
            int u = t - QPN;
            int kv = u / HD, d = u % HD;
            float val = row[t] * QKSCALE;
            if (d < ROTD) {
                float c = cosE[s * ROTD + d], sn = sinE[s * ROTD + d];
                float partner = (d < RHALF) ? -row[QPN + kv * HD + d + RHALF]
                                            :  row[QPN + kv * HD + d - RHALF];
                val = val * c + partner * QKSCALE * sn;
            }
            keys_out[(size_t)kv * HD * SEQ + (size_t)d * SEQ + s] = val;
            kbuf[(size_t)kv * SEQ * HD + (size_t)s * HD + d] = f2bf(val);
        } else {
            int u = t - QPN - KPN;
            int kv = u / HD, d = u % HD;
            float val = row[t];
            vals_out[(size_t)kv * SEQ * HD + (size_t)s * HD + d] = val;
            vT[(size_t)kv * HD * SEQ + (size_t)d * SEQ + s] = f2bf(val);
        }
    }
}

// ---------------------------------------------------------------- flash attention
// One wave32 per (head, 16-query tile). Online softmax, WMMA for QK^T and P*V.
__global__ __launch_bounds__(32) void attn_kernel(
    const unsigned short* __restrict__ qb,
    const unsigned short* __restrict__ kbuf,
    const unsigned short* __restrict__ vT,
    const int* __restrict__ mask_flag,
    unsigned short* __restrict__ aob)  // [SEQ,HIDN] bf16
{
    __shared__ unsigned short pLds[16 * 32];
    const int lane = threadIdx.x;
    const int laneHalf = lane >> 4, laneMod = lane & 15;
    const int head = blockIdx.x / (SEQ / 16);
    const int qt   = blockIdx.x % (SEQ / 16);
    const float maskAdd = -128.0f * (float)(*mask_flag);

    const unsigned short* q = qb   + (size_t)head * SEQ * HD;
    const unsigned short* k = kbuf + (size_t)(head / GQA) * SEQ * HD;
    const unsigned short* v = vT   + (size_t)(head / GQA) * HD * SEQ;

    v16bf qf[4];
    #pragma unroll
    for (int kc = 0; kc < 4; ++kc)
        qf[kc] = ldsFrag(&q[(size_t)(qt * 16 + laneMod) * HD + kc * 32 + laneHalf * 16]);

    v8f oacc[8] = {};
    float mrow[8], lrow[8];
    #pragma unroll
    for (int r = 0; r < 8; ++r) { mrow[r] = -1e30f; lrow[r] = 0.f; }

    for (int kb0 = 0; kb0 < SEQ; kb0 += 32) {
        v8f s[2] = {};
        #pragma unroll
        for (int nt = 0; nt < 2; ++nt)
            #pragma unroll
            for (int kc = 0; kc < 4; ++kc) {
                v16bf kf = ldsFrag(&k[(size_t)(kb0 + nt * 16 + laneMod) * HD + kc * 32 + laneHalf * 16]);
                s[nt] = __builtin_amdgcn_wmma_f32_16x16x32_bf16(
                            false, qf[kc], false, kf, (short)0, s[nt], false, false);
            }

        float scl[8];
        #pragma unroll
        for (int r = 0; r < 8; ++r) {
            int m  = r + 8 * laneHalf;
            int qi = qt * 16 + m;
            #pragma unroll
            for (int nt = 0; nt < 2; ++nt) {
                int ki = kb0 + nt * 16 + laneMod;
                if (ki > qi) s[nt][r] += maskAdd;
            }
            float mx = fmaxf(s[0][r], s[1][r]);
            #pragma unroll
            for (int d = 1; d < 16; d <<= 1) mx = fmaxf(mx, __shfl_xor(mx, d, 32));
            float mnew = fmaxf(mrow[r], mx);
            float sc = expf(mrow[r] - mnew);
            float p0 = expf(s[0][r] - mnew);
            float p1 = expf(s[1][r] - mnew);
            float psum = p0 + p1;
            #pragma unroll
            for (int d = 1; d < 16; d <<= 1) psum += __shfl_xor(psum, d, 32);
            lrow[r] = lrow[r] * sc + psum;
            mrow[r] = mnew;
            scl[r]  = sc;
            s[0][r] = p0; s[1][r] = p1;
        }
        #pragma unroll
        for (int ct = 0; ct < 8; ++ct)
            #pragma unroll
            for (int r = 0; r < 8; ++r) oacc[ct][r] *= scl[r];

        #pragma unroll
        for (int nt = 0; nt < 2; ++nt)
            #pragma unroll
            for (int r = 0; r < 8; ++r)
                pLds[(r + 8 * laneHalf) * 32 + nt * 16 + laneMod] = f2bf(s[nt][r]);
        __syncthreads();
        v16bf pf = ldsFrag(&pLds[laneMod * 32 + laneHalf * 16]);
        #pragma unroll
        for (int ct = 0; ct < 8; ++ct) {
            v16bf vf = ldsFrag(&v[(size_t)(ct * 16 + laneMod) * SEQ + kb0 + laneHalf * 16]);
            oacc[ct] = __builtin_amdgcn_wmma_f32_16x16x32_bf16(
                           false, pf, false, vf, (short)0, oacc[ct], false, false);
        }
        __syncthreads();
    }

    #pragma unroll
    for (int ct = 0; ct < 8; ++ct)
        #pragma unroll
        for (int r = 0; r < 8; ++r) {
            int m  = r + 8 * laneHalf;
            int qi = qt * 16 + m;
            int ch = ct * 16 + laneMod;
            aob[(size_t)qi * HIDN + head * HD + ch] = f2bf(oacc[ct][r] / lrow[r]);
        }
}

// ---------------------------------------------------------------- SiLU(gate)*up
__global__ __launch_bounds__(256) void silu_kernel(
    const float* __restrict__ gu, unsigned short* __restrict__ act)
{
    size_t i = (size_t)blockIdx.x * 256 + threadIdx.x;
    if (i >= (size_t)SEQ * FFN) return;
    size_t s = i / FFN, f = i % FFN;
    float g  = gu[s * 2 * FFN + f];
    float up = gu[s * 2 * FFN + FFN + f];
    float sig = 1.f / (1.f + expf(-g));
    act[i] = f2bf(g * sig * up);
}

// ---------------------------------------------------------------- LM head GEMV
__global__ __launch_bounds__(256) void lmhead_kernel(
    const unsigned short* __restrict__ hl, const float* __restrict__ w,
    float* __restrict__ logits)
{
    int row  = blockIdx.x * 8 + (threadIdx.x >> 5);
    int lane = threadIdx.x & 31;
    const float* wr = w + (size_t)row * HIDN;
    float sum = 0.f;
    for (int c = lane; c < HIDN; c += 32) sum += bf2f(hl[c]) * wr[c];
    #pragma unroll
    for (int d = 1; d < 32; d <<= 1) sum += __shfl_xor(sum, d, 32);
    if (lane == 0) logits[row] = sum;
}

// ---------------------------------------------------------------- argmax + tail
__global__ __launch_bounds__(256) void argmax_kernel(
    const float* __restrict__ logits, int* __restrict__ tail)
{
    __shared__ unsigned long long red[256];
    unsigned long long best = 0ull;
    for (int i = threadIdx.x; i < VOCAB; i += 256) {
        unsigned int u = __float_as_uint(logits[i]);
        unsigned int key = (u & 0x80000000u) ? ~u : (u | 0x80000000u);
        unsigned long long v = ((unsigned long long)key << 32) | (unsigned int)(VOCAB - i);
        best = (v > best) ? v : best;
    }
    red[threadIdx.x] = best;
    __syncthreads();
    for (int st = 128; st > 0; st >>= 1) {
        if (threadIdx.x < st) {
            unsigned long long a = red[threadIdx.x], b = red[threadIdx.x + st];
            red[threadIdx.x] = (a > b) ? a : b;
        }
        __syncthreads();
    }
    if (threadIdx.x == 0) {
        tail[0] = SEQ;                                        // kv_seq_len
        tail[1] = VOCAB - (int)(red[0] & 0xFFFFFFFFull);      // token
    }
}

// ================================================================ launch
extern "C" void kernel_launch(void* const* d_in, const int* in_sizes, int n_in,
                              void* d_out, int out_size, void* d_ws, size_t ws_size,
                              hipStream_t stream)
{
    const int*            input_ids  = (const int*)d_in[0];
    const int*            mask_flag  = (const int*)d_in[1];
    const unsigned char*  embed_q    = (const unsigned char*)d_in[2];
    const float*          embed_sc   = (const float*)d_in[3];
    const float*          embed_z    = (const float*)d_in[4];
    const float*          ln1_w      = (const float*)d_in[5];
    const float*          qkv_w      = (const float*)d_in[6];
    const float*          o_w        = (const float*)d_in[7];
    const float*          ln2_w      = (const float*)d_in[8];
    const float*          gate_up_w  = (const float*)d_in[9];
    const float*          down_w     = (const float*)d_in[10];
    const float*          norm_w     = (const float*)d_in[11];
    const float*          lm_head_w  = (const float*)d_in[12];
    const float*          cosE       = (const float*)d_in[13];
    const float*          sinE       = (const float*)d_in[14];

    // outputs
    float* keys_out = (float*)d_out;                                  // [L,NKV,HD,SEQ]
    float* vals_out = keys_out + (size_t)LAYERS * NKV * HD * SEQ;     // [L,NKV,SEQ,HD]
    int*   tail     = (int*)(vals_out + (size_t)LAYERS * NKV * SEQ * HD);

    // workspace carve
    char* ws = (char*)d_ws;
    float*          h    = (float*)ws;          ws += sizeof(float) * (size_t)SEQ * HIDN;
    unsigned short* hn   = (unsigned short*)ws; ws += 2 * (size_t)SEQ * HIDN;
    float*          qkv  = (float*)ws;          ws += sizeof(float) * (size_t)SEQ * QKVN;
    unsigned short* qb   = (unsigned short*)ws; ws += 2 * (size_t)NH * SEQ * HD;
    unsigned short* kbuf = (unsigned short*)ws; ws += 2 * (size_t)NKV * SEQ * HD;
    unsigned short* vT   = (unsigned short*)ws; ws += 2 * (size_t)NKV * SEQ * HD;
    unsigned short* aob  = (unsigned short*)ws; ws += 2 * (size_t)SEQ * HIDN;
    float*          gu   = (float*)ws;          ws += sizeof(float) * (size_t)SEQ * 2 * FFN;
    unsigned short* act  = (unsigned short*)ws; ws += 2 * (size_t)SEQ * FFN;
    float*          lgts = (float*)ws;          ws += sizeof(float) * (size_t)VOCAB;
    unsigned short* hl   = (unsigned short*)ws; ws += 2 * (size_t)HIDN;

    embed_kernel<<<SEQ, 256, 0, stream>>>(input_ids, embed_q, embed_sc, embed_z, h);

    for (int l = 0; l < LAYERS; ++l) {
        rms_kernel<<<SEQ, 256, 0, stream>>>(h, ln1_w + (size_t)l * HIDN, hn);

        gemm_bf16_kernel<<<dim3(SEQ / BM, QKVN / BN), 256, 0, stream>>>(
            hn, qkv_w + (size_t)l * QKVN * HIDN, nullptr, qkv, SEQ, QKVN, HIDN);

        rope_kv_kernel<<<SEQ, 256, 0, stream>>>(
            qkv, cosE, sinE, qb, kbuf, vT,
            keys_out + (size_t)l * NKV * HD * SEQ,
            vals_out + (size_t)l * NKV * SEQ * HD);

        attn_kernel<<<NH * (SEQ / 16), 32, 0, stream>>>(qb, kbuf, vT, mask_flag, aob);

        gemm_bf16_kernel<<<dim3(SEQ / BM, HIDN / BN), 256, 0, stream>>>(
            aob, o_w + (size_t)l * HIDN * QPN, h, h, SEQ, HIDN, QPN);

        rms_kernel<<<SEQ, 256, 0, stream>>>(h, ln2_w + (size_t)l * HIDN, hn);

        gemm_bf16_kernel<<<dim3(SEQ / BM, 2 * FFN / BN), 256, 0, stream>>>(
            hn, gate_up_w + (size_t)l * 2 * FFN * HIDN, nullptr, gu, SEQ, 2 * FFN, HIDN);

        silu_kernel<<<((size_t)SEQ * FFN + 255) / 256, 256, 0, stream>>>(gu, act);

        gemm_bf16_kernel<<<dim3(SEQ / BM, HIDN / BN), 256, 0, stream>>>(
            act, down_w + (size_t)l * HIDN * FFN, h, h, SEQ, HIDN, FFN);
    }

    rms_kernel<<<1, 256, 0, stream>>>(h + (size_t)(SEQ - 1) * HIDN, norm_w, hl);
    lmhead_kernel<<<VOCAB / 8, 256, 0, stream>>>(hl, lm_head_w, lgts);
    argmax_kernel<<<1, 256, 0, stream>>>(lgts, tail);
}
